// LSTM_23055384445841
// MI455X (gfx1250) — matrix operand order repaired and checked by reading, not compiled, not run on hardware
//
#include <hip/hip_runtime.h>

typedef __attribute__((ext_vector_type(16))) _Float16 v16h;
typedef __attribute__((ext_vector_type(8)))  float    v8f;

#define B_TOT 1024
#define T_LEN 512
#define D_IN  32
#define HID   64

// Fragment layout: each B fragment (32x16 f16, one K-tile x one N-tile) is stored as
// 32 lanes x 16 halves contiguous: [frag][lane][q], q = half index 0..15.
// B element mapping (ISA 7.12.2): lane l, half q -> N = tile*16 + (l&15),
//                                 K = kt*32 + (q>>3)*16 + (l>>4)*8 + (q&7).
// Fragments: 0..15   W_ih0   16..47 W_hh0   48..79 W_ih1   80..111 W_hh1
#define NFRAG        112
#define WPACK_HALVES (NFRAG * 512)      // 57344
#define WPACK_BYTES  (WPACK_HALVES * 2) // 114688
#define BIAS_FLOATS  512                // [0,256)=b_ih0+b_hh0, [256,512)=b_ih1+b_hh1
#define COPY_BYTES   (WPACK_BYTES + BIAS_FLOATS * 4)   // 116736, 16B-multiple
#define STAGE_HALVES 1024               // one 16x64 f16 A-stage
#define SMEM_BYTES   (COPY_BYTES + 3 * STAGE_HALVES * 2) // h1 dbl-buf + h2 = 122880

// ---------------------------------------------------------------------------
// One-time weight repack: f32 row-major -> f16 WMMA B-fragment layout in d_ws.
// ---------------------------------------------------------------------------
__global__ void pack_weights(const float* __restrict__ Wih0, const float* __restrict__ Whh0,
                             const float* __restrict__ bih0, const float* __restrict__ bhh0,
                             const float* __restrict__ Wih1, const float* __restrict__ Whh1,
                             const float* __restrict__ bih1, const float* __restrict__ bhh1,
                             _Float16* __restrict__ wout, float* __restrict__ bout) {
    int idx = blockIdx.x * blockDim.x + threadIdx.x;
    if (idx < WPACK_HALVES) {
        int f   = idx >> 9;        // fragment id
        int rem = idx & 511;
        int l   = rem >> 4;        // lane 0..31
        int q   = rem & 15;        // half 0..15
        int nl  = l & 15, g = l >> 4;
        const float* src; int tile, kt, ld;
        if (f < 16)      { tile = f;                        kt = 0;      src = Wih0; ld = D_IN; }
        else if (f < 48) { int i2 = f - 16; tile = i2 >> 1; kt = i2 & 1; src = Whh0; ld = HID;  }
        else if (f < 80) { int i2 = f - 48; tile = i2 >> 1; kt = i2 & 1; src = Wih1; ld = HID;  }
        else             { int i2 = f - 80; tile = i2 >> 1; kt = i2 & 1; src = Whh1; ld = HID;  }
        int N = tile * 16 + nl;
        int K = kt * 32 + (q >> 3) * 16 + g * 8 + (q & 7);
        wout[idx] = (_Float16)src[N * ld + K];
    } else if (idx < WPACK_HALVES + BIAS_FLOATS) {
        int n = idx - WPACK_HALVES;
        bout[n] = (n < 256) ? (bih0[n] + bhh0[n]) : (bih1[n - 256] + bhh1[n - 256]);
    }
}

// ---------------------------------------------------------------------------
// Fused 2-layer LSTM + FC, software-pipelined across two waves per block:
//   wave 0: layer-0 cell at step k      (produces h1[k])
//   wave 1: layer-1 cell at step k-1    (consumes h1[k-1], produces h2[k-1])
// 64 blocks x 64 threads; each block owns one 16-row batch tile.
// ---------------------------------------------------------------------------
__device__ __forceinline__ v8f wmma16(v16h a, v16h b, v8f c) {
    return __builtin_amdgcn_wmma_f32_16x16x32_f16(false, a, false, b, (short)0, c, false, false);
}
__device__ __forceinline__ float frcp(float x) {
#if __has_builtin(__builtin_amdgcn_rcpf)
    return __builtin_amdgcn_rcpf(x);
#else
    return 1.0f / x;
#endif
}
__device__ __forceinline__ float sigm(float x)  { return frcp(1.0f + __expf(-x)); }
__device__ __forceinline__ float tanha(float x) { return 1.0f - 2.0f * frcp(__expf(2.0f * x) + 1.0f); }

__global__ __launch_bounds__(64)
void lstm_fused(const float* __restrict__ x,
                const _Float16* __restrict__ wpack,   // packed weights+bias, contiguous
                const float* __restrict__ Wfc,
                const float* __restrict__ bfc,
                float* __restrict__ out) {
    extern __shared__ char smem[];
    _Float16* wf    = (_Float16*)smem;
    float*    bias  = (float*)(smem + WPACK_BYTES);
    _Float16* stage = (_Float16*)(smem + COPY_BYTES);  // [0,1024) h1 buf0, [1024,2048) h1 buf1,
                                                       // [2048,3072) h2
    const int tid  = threadIdx.x;
    const int wave = tid >> 5;
    const int lane = tid & 31;
    const int nl   = lane & 15;
    const int gg0  = lane >> 4;
    const int b0   = blockIdx.x * 16;

    // ---- Prologue: async-DMA packed weights+biases into LDS (offset 0..) ----
    // Dynamic LDS starts at workgroup LDS offset 0, so LDS addr == byte index.
    for (int i = tid; i < COPY_BYTES / 16; i += 64) {
        unsigned int loff = (unsigned int)(i * 16);
        const char* gaddr = (const char*)wpack + (size_t)i * 16;
        asm volatile("global_load_async_to_lds_b128 %0, %1, off"
                     :: "v"(loff), "v"(gaddr) : "memory");
    }
    // Zero the three h stages with plain LDS stores.
    {
        uint4 z; z.x = 0; z.y = 0; z.z = 0; z.w = 0;
        uint4* st = (uint4*)stage;
        for (int i = tid; i < (3 * STAGE_HALVES * 2) / 16; i += 64) st[i] = z;
    }
    asm volatile("s_wait_asynccnt 0x0" ::: "memory");
    __syncthreads();

    const v16h* W  = (const v16h*)wf;     // W[frag*32 + lane]
    const v16h* Sv = (const v16h*)stage;  // stage viewed as v16h[96*...]: 64 per h-buffer

    // Per-lane A-stage write base for hidden column n = j*16+nl (ISA A-frag layout).
    int sbase[4];
    for (int j = 0; j < 4; ++j) {
        int n  = j * 16 + nl;
        int kt = n >> 5, kk = n & 31;
        int ga = (kk >> 3) & 1;
        int q  = (kk & 7) + ((kk >> 4) << 3);
        sbase[j] = kt * 512 + (8 * gg0 + 16 * ga) * 16 + q;  // + r*16 per row
    }

    // Persistent splatted biases for this wave's layer (C operand of first WMMA
    // in each gate chain -> no per-step accumulator-init movs).
    v8f bI[4], bF[4], bG[4], bO[4];
    {
        int boff = (wave == 1) ? 256 : 0;
        for (int j = 0; j < 4; ++j) {
            int n = j * 16 + nl;
            float bi = bias[boff + n],       bf = bias[boff + 64 + n];
            float bg = bias[boff + 128 + n], bo = bias[boff + 192 + n];
            for (int r = 0; r < 8; ++r) { bI[j][r] = bi; bF[j][r] = bf; bG[j][r] = bg; bO[j][r] = bo; }
        }
    }

    v8f cst[4], fcacc;                    // c-state for this wave's layer
    for (int j = 0; j < 4; ++j)
        for (int r = 0; r < 8; ++r) cst[j][r] = 0.f;
    for (int r = 0; r < 8; ++r) fcacc[r] = 0.f;

    float wfc_l[4];
    for (int j = 0; j < 4; ++j) wfc_l[j] = Wfc[j * 16 + nl];

    const float* xrow = x + (size_t)(b0 + nl) * (T_LEN * D_IN);

    for (int k = 0; k <= T_LEN; ++k) {
        // Integer element offsets into the single LDS stage base (keeps the
        // accesses inferable as LDS -> ds_load/ds_store, not flat).
        const int rsel = ((k + 1) & 1) * 64;   // v16h index of h1[k-1] buffer
        const int wsel = (k & 1) * STAGE_HALVES;

        if (wave == 0 && k < T_LEN) {
            // ---------------- layer 0, step k ----------------
            const float* xb = xrow + k * D_IN;
            if (k + 1 < T_LEN) __builtin_prefetch(xrow + (k + 1) * D_IN, 0, 1);

            float4 a0 = *(const float4*)(xb + gg0 * 8);
            float4 a1 = *(const float4*)(xb + gg0 * 8 + 4);
            float4 a2 = *(const float4*)(xb + 16 + gg0 * 8);
            float4 a3 = *(const float4*)(xb + 20 + gg0 * 8);
            v16h ax;
            ax[0]  = (_Float16)a0.x; ax[1]  = (_Float16)a0.y; ax[2]  = (_Float16)a0.z; ax[3]  = (_Float16)a0.w;
            ax[4]  = (_Float16)a1.x; ax[5]  = (_Float16)a1.y; ax[6]  = (_Float16)a1.z; ax[7]  = (_Float16)a1.w;
            ax[8]  = (_Float16)a2.x; ax[9]  = (_Float16)a2.y; ax[10] = (_Float16)a2.z; ax[11] = (_Float16)a2.w;
            ax[12] = (_Float16)a3.x; ax[13] = (_Float16)a3.y; ax[14] = (_Float16)a3.z; ax[15] = (_Float16)a3.w;

            v16h aha = Sv[rsel + lane], ahb = Sv[rsel + 32 + lane];   // h1[k-1]

#pragma unroll
            for (int j = 0; j < 4; ++j) {
                v8f gi, gf, gG, go;
                gi = wmma16(ax, W[(0  + j) * 32 + lane], bI[j]);   // C = bias, D = fresh acc
                gf = wmma16(ax, W[(4  + j) * 32 + lane], bF[j]);
                gG = wmma16(ax, W[(8  + j) * 32 + lane], bG[j]);
                go = wmma16(ax, W[(12 + j) * 32 + lane], bO[j]);
                gi = wmma16(aha, W[(16 + (0  + j) * 2 + 0) * 32 + lane], gi);
                gi = wmma16(ahb, W[(16 + (0  + j) * 2 + 1) * 32 + lane], gi);
                gf = wmma16(aha, W[(16 + (4  + j) * 2 + 0) * 32 + lane], gf);
                gf = wmma16(ahb, W[(16 + (4  + j) * 2 + 1) * 32 + lane], gf);
                gG = wmma16(aha, W[(16 + (8  + j) * 2 + 0) * 32 + lane], gG);
                gG = wmma16(ahb, W[(16 + (8  + j) * 2 + 1) * 32 + lane], gG);
                go = wmma16(aha, W[(16 + (12 + j) * 2 + 0) * 32 + lane], go);
                go = wmma16(ahb, W[(16 + (12 + j) * 2 + 1) * 32 + lane], go);
#pragma unroll
                for (int r = 0; r < 8; ++r) {
                    float iv = sigm(gi[r]), fv = sigm(gf[r]);
                    float gv = tanha(gG[r]), ov = sigm(go[r]);
                    float c = fv * cst[j][r] + iv * gv;
                    cst[j][r] = c;
                    float h = ov * tanha(c);
                    stage[wsel + sbase[j] + r * 16] = (_Float16)h;   // h1[k]
                }
            }
        }

        if (wave == 1 && k > 0) {
            // ---------------- layer 1, step k-1 ----------------
            v16h ah1a = Sv[rsel + lane],      ah1b = Sv[rsel + 32 + lane];  // h1[k-1]
            v16h ah2a = Sv[128 + lane],       ah2b = Sv[128 + 32 + lane];   // h2[k-2]

#pragma unroll
            for (int j = 0; j < 4; ++j) {
                v8f gi, gf, gG, go;
                gi = wmma16(ah1a, W[(48 + (0  + j) * 2 + 0) * 32 + lane], bI[j]); // C = bias
                gf = wmma16(ah1a, W[(48 + (4  + j) * 2 + 0) * 32 + lane], bF[j]);
                gG = wmma16(ah1a, W[(48 + (8  + j) * 2 + 0) * 32 + lane], bG[j]);
                go = wmma16(ah1a, W[(48 + (12 + j) * 2 + 0) * 32 + lane], bO[j]);
                gi = wmma16(ah1b, W[(48 + (0  + j) * 2 + 1) * 32 + lane], gi);
                gf = wmma16(ah1b, W[(48 + (4  + j) * 2 + 1) * 32 + lane], gf);
                gG = wmma16(ah1b, W[(48 + (8  + j) * 2 + 1) * 32 + lane], gG);
                go = wmma16(ah1b, W[(48 + (12 + j) * 2 + 1) * 32 + lane], go);
                gi = wmma16(ah2a, W[(80 + (0  + j) * 2 + 0) * 32 + lane], gi);
                gi = wmma16(ah2b, W[(80 + (0  + j) * 2 + 1) * 32 + lane], gi);
                gf = wmma16(ah2a, W[(80 + (4  + j) * 2 + 0) * 32 + lane], gf);
                gf = wmma16(ah2b, W[(80 + (4  + j) * 2 + 1) * 32 + lane], gf);
                gG = wmma16(ah2a, W[(80 + (8  + j) * 2 + 0) * 32 + lane], gG);
                gG = wmma16(ah2b, W[(80 + (8  + j) * 2 + 1) * 32 + lane], gG);
                go = wmma16(ah2a, W[(80 + (12 + j) * 2 + 0) * 32 + lane], go);
                go = wmma16(ah2b, W[(80 + (12 + j) * 2 + 1) * 32 + lane], go);
#pragma unroll
                for (int r = 0; r < 8; ++r) {
                    float iv = sigm(gi[r]), fv = sigm(gf[r]);
                    float gv = tanha(gG[r]), ov = sigm(go[r]);
                    float c = fv * cst[j][r] + iv * gv;
                    cst[j][r] = c;
                    float h = ov * tanha(c);
                    stage[2 * STAGE_HALVES + sbase[j] + r * 16] = (_Float16)h;  // h2[k-1]
                    if (k == T_LEN) fcacc[r] += h * wfc_l[j];   // h2[T-1] -> FC
                }
            }
        }
        __syncthreads();
    }

    // Final FC (wave 1): reduce partial dot over the 16 lanes sharing each row.
    if (wave == 1) {
        float bfc0 = bfc[0];
#pragma unroll
        for (int r = 0; r < 8; ++r) {
            float v = fcacc[r];
            v += __shfl_xor(v, 1, 32);
            v += __shfl_xor(v, 2, 32);
            v += __shfl_xor(v, 4, 32);
            v += __shfl_xor(v, 8, 32);
            if (nl == 0) out[b0 + gg0 * 8 + r] = v + bfc0;
        }
    }
}

// ---------------------------------------------------------------------------
extern "C" void kernel_launch(void* const* d_in, const int* in_sizes, int n_in,
                              void* d_out, int out_size, void* d_ws, size_t ws_size,
                              hipStream_t stream) {
    const float* x    = (const float*)d_in[0];
    const float* Wih0 = (const float*)d_in[1];
    const float* Whh0 = (const float*)d_in[2];
    const float* bih0 = (const float*)d_in[3];
    const float* bhh0 = (const float*)d_in[4];
    const float* Wih1 = (const float*)d_in[5];
    const float* Whh1 = (const float*)d_in[6];
    const float* bih1 = (const float*)d_in[7];
    const float* bhh1 = (const float*)d_in[8];
    const float* Wfc  = (const float*)d_in[9];
    const float* bfc  = (const float*)d_in[10];

    _Float16* wpack = (_Float16*)d_ws;
    float*    bpack = (float*)((char*)d_ws + WPACK_BYTES);

    int total = WPACK_HALVES + BIAS_FLOATS;
    pack_weights<<<(total + 255) / 256, 256, 0, stream>>>(
        Wih0, Whh0, bih0, bhh0, Wih1, Whh1, bih1, bhh1, wpack, bpack);

    (void)hipFuncSetAttribute(reinterpret_cast<const void*>(lstm_fused),
                              hipFuncAttributeMaxDynamicSharedMemorySize, SMEM_BYTES);
    lstm_fused<<<B_TOT / 16, 64, SMEM_BYTES, stream>>>(
        x, wpack, Wfc, bfc, (float*)d_out);
}
